// LanguageModelRNN_20813411517104
// MI455X (gfx1250) — matrix operand order repaired and checked by reading, not compiled, not run on hardware
//
#include <hip/hip_runtime.h>
#include <hip/hip_bf16.h>

// ---------------------------------------------------------------------------
// LanguageModelRNN for MI455X (gfx1250, wave32, WMMA + async-to-LDS + TDM)
//   B=4, T=512, V=32000, E=256, H=512, 3H=1536, BT=2048
//   Pipeline: transpose-convert weights->f16, gather embeddings (f16),
//             WMMA GEMM xp = xe@W + b,
//             GRU recurrence (32 WGPs, U-slices staged by TDM into LDS),
//             WMMA GEMM logits = hs@Wo + bo (dominant 67 GFLOP).
//   GEMMs: A tile via GLOBAL_LOAD_ASYNC_TO_LDS_B128 (ASYNCcnt),
//          B tile via TENSOR_LOAD_TO_LDS (TENSORcnt, wave0-issued, LDS pad
//          feature reproduces the 80B padded row stride), double-buffered.
// ---------------------------------------------------------------------------

typedef __attribute__((ext_vector_type(16))) _Float16 v16h;
typedef __attribute__((ext_vector_type(8)))  _Float16 v8h;
typedef __attribute__((ext_vector_type(8)))  float    v8f;
typedef __attribute__((ext_vector_type(4)))  unsigned int u32x4;
typedef __attribute__((ext_vector_type(8)))  int i32x8;
typedef __attribute__((ext_vector_type(4)))  int i32x4;

#define BB   4
#define TT   512
#define VV   32000
#define EE   256
#define HH   512
#define H3   1536
#define BT   2048      // B*T
#define GRU_NWG 32
#define GRU_THREADS 128

#if defined(__has_builtin)
#if __has_builtin(__builtin_amdgcn_tensor_load_to_lds) && \
    __has_builtin(__builtin_amdgcn_s_wait_tensorcnt)
#define HAVE_TDM 1
#endif
#endif

#if defined(HAVE_TDM)
#define ASYNC_OPS_PER_TILE 1   // A tile only (B tile goes through TDM)
#else
#define ASYNC_OPS_PER_TILE 3   // A tile (1) + B tile (2)
#endif

// ---------------------------------------------------------------------------
// CDNA5 async copy helpers (ISA 15.18.3, tracked by ASYNCcnt)
// ---------------------------------------------------------------------------
__device__ __forceinline__ void async_ld_b128(const void* gptr, unsigned lds_off) {
    asm volatile("global_load_async_to_lds_b128 %0, %1, off"
                 :: "v"(lds_off), "v"(gptr) : "memory");
}
template <int N>
__device__ __forceinline__ void wait_asynccnt() {
    asm volatile("s_wait_asynccnt %0" :: "i"(N) : "memory");
}

#if defined(HAVE_TDM)
// ---------------------------------------------------------------------------
// Tensor Data Mover: 2D tile load, D# per CDNA5 ISA.
//   tile_w elems/row (f16), tile_rows rows, source row stride row_stride
//   elems, tensor_rows total rows (OOB bound), LDS pad codes:
//   pad_int: DWORDs-before-pad code (2->8DW, 3->16DW), pad_amt: DWORDs-1.
// This toolchain exposes the 6-arg builtin (clang-23 form):
//   (uint32x4 g0, int32x8 g1, int32x4, int32x4, int32x8, i32 cpol)
// ---------------------------------------------------------------------------
__device__ __forceinline__ void tdm_load_2d(const void* gptr, unsigned lds_addr,
                                            unsigned tile_w, unsigned tile_rows,
                                            unsigned row_stride, unsigned tensor_rows,
                                            unsigned pad_int, unsigned pad_amt) {
    unsigned long long ga = (unsigned long long)(uintptr_t)gptr;
    u32x4 g0;
    g0[0] = 1u;                                    // count=1, no gather
    g0[1] = lds_addr;                              // lds_addr [63:32]
    g0[2] = (unsigned)ga;                          // global_addr lo
    g0[3] = (unsigned)((ga >> 32) & 0x01FFFFFFull) // global_addr [56:32]
            | (2u << 30);                          // type=2 ("image")
    i32x8 g1;
    g1[0] = (int)((1u << 16)                       // data_size=1 -> 2 bytes
                  | (1u << 20)                     // pad_enable
                  | (pad_int << 22)                // pad_interval
                  | (pad_amt << 25));              // pad_amount
    g1[1] = (int)((row_stride & 0xFFFFu) << 16);   // tensor_dim0[15:0]
    g1[2] = (int)(((tensor_rows & 0xFFFFu) << 16)  // tensor_dim1[15:0]
                  | ((row_stride >> 16) & 0xFFFFu));
    g1[3] = (int)(((tile_w & 0xFFFFu) << 16)       // tile_dim0
                  | ((tensor_rows >> 16) & 0xFFFFu));
    g1[4] = (int)(tile_rows & 0xFFFFu);            // tile_dim1 (tile_dim2=0)
    g1[5] = (int)row_stride;                       // tensor_dim0_stride lo32
    g1[6] = 0;
    g1[7] = 0;
    i32x4 z4 = {0, 0, 0, 0};                       // 2D: groups 2/3 disabled
    i32x8 z8 = {0, 0, 0, 0, 0, 0, 0, 0};
    __builtin_amdgcn_tensor_load_to_lds(g0, g1, z4, z4, z8, 0);
}
#endif

// ---------------------------------------------------------------------------
// Utility kernels
// ---------------------------------------------------------------------------
__global__ void f32_to_f16_kernel(const float* __restrict__ src,
                                  _Float16* __restrict__ dst, int n) {
    int stride = gridDim.x * blockDim.x;
    for (int i = blockIdx.x * blockDim.x + threadIdx.x; i < n; i += stride)
        dst[i] = (_Float16)src[i];
}

// dst[c][r] = (f16)src[r][c];  src is R x C (f32), dst is C x R (f16).
__global__ __launch_bounds__(256)
void transpose_f32_to_f16_kernel(const float* __restrict__ src,
                                 _Float16* __restrict__ dst, int R, int C) {
    __shared__ float tile[32][33];
    const int tx = threadIdx.x;
    const int ty = threadIdx.y;
    const int c0 = blockIdx.x * 32;
    const int r0 = blockIdx.y * 32;
#pragma unroll
    for (int i = 0; i < 32; i += 8)
        tile[ty + i][tx] = src[(size_t)(r0 + ty + i) * C + c0 + tx];
    __syncthreads();
#pragma unroll
    for (int i = 0; i < 32; i += 8)
        dst[(size_t)(c0 + ty + i) * R + r0 + tx] = (_Float16)tile[tx][ty + i];
}

__global__ void gather_embed_kernel(const int* __restrict__ x,
                                    const float* __restrict__ emb,
                                    _Float16* __restrict__ xe) {
    int stride = gridDim.x * blockDim.x;
    for (int i = blockIdx.x * blockDim.x + threadIdx.x; i < BT * EE; i += stride) {
        int m = i / EE;
        int e = i - m * EE;
        int tok = x[m];
        xe[i] = (_Float16)emb[(size_t)tok * EE + e];
    }
}

__global__ void init_state_kernel(float* __restrict__ hbuf,
                                  unsigned* __restrict__ cnt) {
    for (int i = blockIdx.x * blockDim.x + threadIdx.x; i < 2 * BB * HH;
         i += gridDim.x * blockDim.x)
        hbuf[i] = 0.0f;
    if (blockIdx.x == 0 && threadIdx.x == 0) *cnt = 0u;
}

// ---------------------------------------------------------------------------
// Tiled f16 WMMA GEMM:  C[M,N] (f32) = A[M,K] x Bt[N,K]^T + bias[N]
// Workgroup tile 64x128, 8 waves, wave tile 16x64 (4 accumulators),
// double-buffered LDS; A via async copies, B via TDM (wave0).
// Requires M%64==0, N%128==0, K%32==0.
// ---------------------------------------------------------------------------
__device__ __forceinline__ void stage_tile(
    const _Float16* __restrict__ A, const _Float16* __restrict__ Bt,
    int K, int N, int m0, int n0, int kb, int tid,
    _Float16 (*As)[40], _Float16 (*Bs)[40]) {
    // A tile 64x32: one b128 async chunk per thread (1 op per wave)
    {
        int o   = tid * 8;
        int row = o >> 5;
        int col = o & 31;
        async_ld_b128(A + (size_t)(m0 + row) * K + kb + col,
                      (unsigned)(uintptr_t)&As[row][col]);
    }
#if defined(HAVE_TDM)
    // B tile 128x32 via one TDM descriptor; wave 0 only (uniform branch so
    // other waves never fetch the instruction -- TDM ignores EXEC).
    if (__builtin_amdgcn_readfirstlane(tid) == 0) {
        tdm_load_2d(Bt + (size_t)n0 * K + kb, (unsigned)(uintptr_t)&Bs[0][0],
                    /*tile_w=*/32u, /*tile_rows=*/128u,
                    /*row_stride=*/(unsigned)K, /*tensor_rows=*/(unsigned)N,
                    /*pad_int: 16 DW=64B*/3u, /*pad_amt: 4 DW=16B*/3u);
    }
#else
#pragma unroll
    for (int it = 0; it < 2; ++it) {
        int c   = tid + it * 256;
        int row = c >> 2;
        int kc  = (c & 3) * 8;
        async_ld_b128(Bt + (size_t)(n0 + row) * K + kb + kc,
                      (unsigned)(uintptr_t)&Bs[row][kc]);
    }
#endif
}

__global__ __launch_bounds__(256)
void gemm_f16_wmma_kernel(const _Float16* __restrict__ A,
                          const _Float16* __restrict__ Bt,
                          const float* __restrict__ bias,
                          float* __restrict__ C,
                          int M, int N, int K) {
    // Row stride 40 f16 = 80 B (16B multiple; 64B data + 16B pad == TDM pad).
    __shared__ __align__(16) _Float16 As[2][64][40];
    __shared__ __align__(16) _Float16 Bs[2][128][40];

    const int tid   = threadIdx.x;
    const int lane  = tid & 31;
    const int wid   = tid >> 5;
    const int mtile = wid & 3;
    const int nquad = wid >> 2;
    const int lr    = lane & 15;
    const int lh    = lane >> 4;

    const int m0 = blockIdx.y * 64;
    const int n0 = blockIdx.x * 128;

    const v8f vzero = {0.f, 0.f, 0.f, 0.f, 0.f, 0.f, 0.f, 0.f};
    v8f acc[4];
#pragma unroll
    for (int i = 0; i < 4; ++i) acc[i] = vzero;

    const int am = mtile * 16 + lr;
    const int nk = K / 32;

    stage_tile(A, Bt, K, N, m0, n0, 0, tid, As[0], Bs[0]);

    for (int ki = 0; ki < nk; ++ki) {
        const int cur = ki & 1;
        if (ki + 1 < nk) {
            stage_tile(A, Bt, K, N, m0, n0, (ki + 1) * 32, tid,
                       As[cur ^ 1], Bs[cur ^ 1]);
            wait_asynccnt<ASYNC_OPS_PER_TILE>();
#if defined(HAVE_TDM)
            __builtin_amdgcn_s_wait_tensorcnt(1);
#endif
        } else {
            wait_asynccnt<0>();
#if defined(HAVE_TDM)
            __builtin_amdgcn_s_wait_tensorcnt(0);
#endif
        }
        __syncthreads();

        // ---- A fragment (ISA 16-bit A 16x32 layout) ----
        v8h alo = *(const v8h*)&As[cur][am][lh * 8];
        v8h ahi = *(const v8h*)&As[cur][am][16 + lh * 8];
        v16h afrag = __builtin_shufflevector(alo, ahi,
                        0,1,2,3,4,5,6,7,8,9,10,11,12,13,14,15);

        // ---- 4 B fragments + WMMAs ----
#pragma unroll
        for (int i = 0; i < 4; ++i) {
            int bn = (nquad * 4 + i) * 16 + lr;
            v8h blo = *(const v8h*)&Bs[cur][bn][lh * 8];
            v8h bhi = *(const v8h*)&Bs[cur][bn][16 + lh * 8];
            v16h bfrag = __builtin_shufflevector(blo, bhi,
                            0,1,2,3,4,5,6,7,8,9,10,11,12,13,14,15);
            acc[i] = __builtin_amdgcn_wmma_f32_16x16x32_f16(
                false, afrag, false, bfrag, (short)0, acc[i], false, false);
        }
        __syncthreads();
    }

    // ---- epilogue: add bias, store f32 ----
#pragma unroll
    for (int i = 0; i < 4; ++i) {
        int n = n0 + (nquad * 4 + i) * 16 + lr;
        float bv = bias[n];
#pragma unroll
        for (int r = 0; r < 8; ++r) {
            int m = m0 + mtile * 16 + lh * 8 + r;
            C[(size_t)m * N + n] = acc[i][r] + bv;
        }
    }
    (void)M;
}

// ---------------------------------------------------------------------------
// GRU recurrence. 32 workgroups x 128 threads; each WG's U columns
// (512 x [z16|r16|h16] f16 = 48KB) staged into LDS by three TDM descriptors
// whose pad skip (32B data + 64B skip per row) interleaves the gates.
// ---------------------------------------------------------------------------
__global__ __launch_bounds__(GRU_THREADS)
void gru_kernel(const _Float16* __restrict__ U16,   // [512][1536] f16
                const float* __restrict__ xp,       // [BT][1536] f32
                const int* __restrict__ x,          // [B][T]
                float* __restrict__ hbuf,           // [2][B][H] f32
                _Float16* __restrict__ hs16,        // [BT][H] f16
                unsigned* __restrict__ cnt) {
    __shared__ __align__(16) _Float16 Us[512][48];

    const int wg  = blockIdx.x;
    const int tid = threadIdx.x;

#if defined(HAVE_TDM)
    if (__builtin_amdgcn_readfirstlane(tid) == 0) {
#pragma unroll
        for (int g = 0; g < 3; ++g) {
            tdm_load_2d(U16 + g * HH + wg * 16,
                        (unsigned)(uintptr_t)&Us[0][0] + g * 32u,
                        /*tile_w=*/16u, /*tile_rows=*/512u,
                        /*row_stride=*/(unsigned)H3, /*tensor_rows=*/512u,
                        /*pad_int: 8 DW=32B*/2u, /*pad_amt: 16 DW=64B*/15u);
        }
    }
    __builtin_amdgcn_s_wait_tensorcnt(0);
#else
    for (int idx = tid; idx < 512 * 48; idx += GRU_THREADS) {
        int k  = idx / 48;
        int c  = idx - k * 48;
        int g  = c >> 4;
        int jl = c & 15;
        Us[k][c] = U16[(size_t)k * H3 + g * HH + wg * 16 + jl];
    }
#endif
    __syncthreads();

    const int p    = tid >> 1;         // pair 0..63  (batch x jloc)
    const int kh   = tid & 1;          // K half
    const int b    = p >> 4;
    const int jloc = p & 15;
    const int j    = wg * 16 + jloc;

    for (int t = 0; t < TT; ++t) {
        volatile const float* hc = hbuf + (t & 1) * (BB * HH) + b * HH;
        float s0 = 0.f, s1 = 0.f, s2 = 0.f;
        const int k0 = kh * 256;
#pragma unroll 4
        for (int k = k0; k < k0 + 256; ++k) {
            float hk = hc[k];
            s0 += hk * (float)Us[k][jloc];
            s1 += hk * (float)Us[k][16 + jloc];
            s2 += hk * (float)Us[k][32 + jloc];
        }
        s0 += __shfl_xor(s0, 1);
        s1 += __shfl_xor(s1, 1);
        s2 += __shfl_xor(s2, 1);

        if (kh == 0) {
            int m = b * TT + t;
            const float* xr = xp + (size_t)m * H3;
            float xz = xr[j], xrr = xr[HH + j], xh = xr[2 * HH + j];
            float z  = 1.f / (1.f + __expf(-(xz + s0)));
            float r  = 1.f / (1.f + __expf(-(xrr + s1)));
            float hcand = tanhf(xh + r * s2);
            float hold  = hc[j];
            float hn = z * hold + (1.f - z) * hcand;
            if (x[b * TT + t] == 0) hn = hold;   // mask_zero
            hbuf[((t + 1) & 1) * (BB * HH) + b * HH + j] = hn;
            hs16[(size_t)m * HH + j] = (_Float16)hn;
        }

        // ---- device-wide step barrier (monotone counter) ----
        __threadfence();
        __syncthreads();
        if (tid == 0) {
            atomicAdd(cnt, 1u);
            unsigned target = (unsigned)(GRU_NWG * (t + 1));
            while (atomicAdd(cnt, 0u) < target) {
                __builtin_amdgcn_s_sleep(1);
            }
        }
        __syncthreads();
    }
}

// ---------------------------------------------------------------------------
// Host-side launch
// ---------------------------------------------------------------------------
static inline size_t align_up(size_t v, size_t a) { return (v + a - 1) & ~(a - 1); }

extern "C" void kernel_launch(void* const* d_in, const int* in_sizes, int n_in,
                              void* d_out, int out_size, void* d_ws, size_t ws_size,
                              hipStream_t stream) {
    const int*   x   = (const int*)d_in[0];
    const float* emb = (const float*)d_in[1];
    const float* W   = (const float*)d_in[2];
    const float* U   = (const float*)d_in[3];
    const float* b   = (const float*)d_in[4];
    const float* Wo  = (const float*)d_in[5];
    const float* bo  = (const float*)d_in[6];
    float* out = (float*)d_out;

    char* ws = (char*)d_ws;
    size_t off = 0;
    auto take = [&](size_t bytes) -> char* {
        char* p = ws + off;
        off = align_up(off + bytes, 256);
        return p;
    };
    _Float16* Wo16T = (_Float16*)take((size_t)VV * HH * 2);  // [32000][512]
    _Float16* W16T  = (_Float16*)take((size_t)H3 * EE * 2);  // [1536][256]
    _Float16* U16   = (_Float16*)take((size_t)HH * H3 * 2);  // [512][1536]
    _Float16* xe16  = (_Float16*)take((size_t)BT * EE * 2);
    _Float16* hs16  = (_Float16*)take((size_t)BT * HH * 2);
    float*    xp    = (float*)take((size_t)BT * H3 * 4);
    float*    hbuf  = (float*)take((size_t)2 * BB * HH * 4);
    unsigned* cnt   = (unsigned*)take(256);

    init_state_kernel<<<4, 256, 0, stream>>>(hbuf, cnt);

    f32_to_f16_kernel<<<1024, 256, 0, stream>>>(U, U16, HH * H3);
    transpose_f32_to_f16_kernel<<<dim3(H3 / 32, EE / 32), dim3(32, 8), 0, stream>>>(
        W, W16T, EE, H3);
    transpose_f32_to_f16_kernel<<<dim3(VV / 32, HH / 32), dim3(32, 8), 0, stream>>>(
        Wo, Wo16T, HH, VV);

    gather_embed_kernel<<<2048, 256, 0, stream>>>(x, emb, xe16);

    gemm_f16_wmma_kernel<<<dim3(H3 / 128, BT / 64), 256, 0, stream>>>(
        xe16, W16T, b, xp, BT, H3, EE);

    gru_kernel<<<GRU_NWG, GRU_THREADS, 0, stream>>>(U16, xp, x, hbuf, hs16, cnt);

    gemm_f16_wmma_kernel<<<dim3(VV / 128, BT / 64), 256, 0, stream>>>(
        hs16, Wo16T, bo, out, BT, VV, HH);

    (void)in_sizes; (void)n_in; (void)out_size; (void)ws_size;
}
